// CausalGRUModel_39745627357265
// MI455X (gfx1250) — compile-verified, matrix operand order checked by reading, not run on hardware
//
#include <hip/hip_runtime.h>
#include <hip/hip_bf16.h>

// ---------------------------------------------------------------------------
// Persistent LayerNorm-GRU for MI455X (gfx1250, wave32, WMMA).
//   B=16 (one WMMA M-tile), H=D=512, 3H=1536 gate cols, T=1500, L=2.
//   - bf16 WMMA 16x16x32, f32 accumulate; ds_load_b128 fragment feeds
//   - weights pre-packed once into LDS in WMMA-B layout (96KB/WGP, resident)
//   - N split over 64 workgroups, K split over 6 waves + b128 LDS reduce
//   - TDM (tensor_load_to_lds) stages the recurrent A matrix each step,
//     fenced with s_wait_tensorcnt
//   - 2 device barriers per layer-step (monotone-counter barrier + cluster
//     barrier fast path)
// ---------------------------------------------------------------------------

#define T_STEPS   1500
#define BATCH     16
#define HDIM      512
#define NGATE     1536
#define NBLK      64            // persistent blocks (<= #WGPs on MI455X-class)
#define NTHREADS  256           // 8 wave32
#define NTILES    3             // 48 N-cols per block
#define SLICE_COLS 48

// LDS layout (dynamic): byte offsets
#define LDS_WPACK   0                       // 2*3*16*32*8 dwords = 98304 B
#define LDS_ABUF    98304                   // 16x512 bf16 (ushort) = 16384 B
#define LDS_PART    (98304 + 16384)         // 6*32*8 f32 = 6144 B
#define LDS_RED     (98304 + 16384 + 6144)  // 256 float2 = 2048 B
#define LDS_BYTES   (98304 + 16384 + 6144 + 2048)   // 122880 B

typedef __attribute__((ext_vector_type(16))) __bf16 v16bf;
typedef __attribute__((ext_vector_type(8)))  float  v8f;
typedef __attribute__((ext_vector_type(4)))  unsigned v4u;
typedef __attribute__((ext_vector_type(8)))  int      v8i;
typedef __attribute__((ext_vector_type(4)))  int      v4i;

union Frag16 { v16bf v; uint4 q[2]; unsigned u[8]; };

__device__ __forceinline__ unsigned short f32_to_bf16(float f) {
  unsigned u = __float_as_uint(f);
  unsigned r = (u + 0x7FFFu + ((u >> 16) & 1u)) >> 16;   // round-nearest-even
  return (unsigned short)r;
}

__device__ __forceinline__ float sigmf(float x) { return 1.f / (1.f + __expf(-x)); }

// 1-D contiguous TDM copy: global (2-byte elements) -> LDS.
// D# built per CDNA5 ISA 8.3/8.4: group0 = {count, lds_addr, global_addr, type},
// group1 = {data_size, tensor_dim0, tensor_dim1=1, tile_dim0, strides}.
// 6-arg builtin form (this toolchain): (v4u, v8i, v4i, v4i, v8i, i32 cpol).
__device__ __forceinline__ void tdm_copy_to_lds(const void* gsrc, unsigned lds_off,
                                                unsigned nelem) {
  const unsigned long long ga = (unsigned long long)(uintptr_t)gsrc;
  v4u g0;
  g0[0] = 1u;                                   // count=1 (valid user descriptor)
  g0[1] = lds_off;                              // lds_addr (bytes)
  g0[2] = (unsigned)(ga & 0xffffffffu);         // global_addr[31:0]
  g0[3] = (unsigned)(ga >> 32) | (2u << 30);    // global_addr[56:32] | type=2
  v8i g1;
  g1[0] = (int)(1u << 16);                      // workgroup_mask=0, data_size=1 (2B)
  g1[1] = (int)(nelem << 16);                   // tensor_dim0 low16 -> bits 63:48
  g1[2] = (int)((nelem >> 16) | (1u << 16));    // tensor_dim0 hi | tensor_dim1=1
  g1[3] = (int)(nelem << 16);                   // tile_dim0 -> bits 127:112
  g1[4] = 0;                                    // tile_dim1=0, tile_dim2=0
  g1[5] = (int)nelem;                           // tensor_dim0_stride low
  g1[6] = 0;
  g1[7] = 0;
  const v4i gz4 = {0, 0, 0, 0};
  const v8i gz8 = {0, 0, 0, 0, 0, 0, 0, 0};
  __builtin_amdgcn_tensor_load_to_lds(g0, g1, gz4, gz4, gz8, 0);
}

// Monotone-counter device barrier: barrier i complete when cnt >= NBLK*i.
__device__ __forceinline__ void grid_barrier(unsigned* cnt, unsigned& bi) {
  __syncthreads();
  __builtin_amdgcn_s_cluster_barrier();
  ++bi;
  if (threadIdx.x == 0) {
    __threadfence();                       // release: staging/state stores
    atomicAdd(cnt, 1u);
    const unsigned tgt = (unsigned)NBLK * bi;
    while (__hip_atomic_load(cnt, __ATOMIC_RELAXED, __HIP_MEMORY_SCOPE_AGENT) < tgt)
      __builtin_amdgcn_s_sleep(2);
    __threadfence();                       // acquire
  }
  __syncthreads();
}

__device__ __forceinline__ float2 block_reduce2(float s, float q, float2* red) {
  const int tid = threadIdx.x;
  red[tid] = make_float2(s, q);
  __syncthreads();
  #pragma unroll
  for (int off = 128; off > 0; off >>= 1) {
    if (tid < off) {
      float2 a = red[tid], b = red[tid + off];
      red[tid] = make_float2(a.x + b.x, a.y + b.y);
    }
    __syncthreads();
  }
  float2 r = red[0];
  __syncthreads();
  return r;
}

__global__ void gru_init_39745627357265(unsigned* cnt, float* hf, unsigned short* hb) {
  if (blockIdx.x == 0 && threadIdx.x == 0) *cnt = 0u;
  for (int s = threadIdx.x + blockIdx.x * blockDim.x; s < 2 * BATCH * HDIM;
       s += blockDim.x * gridDim.x) {
    hf[s] = 0.f;
    hb[s] = 0;
  }
}

__global__ __launch_bounds__(NTHREADS)
void gru_persistent_39745627357265(
    const float* __restrict__ x,    // [16,1500,512]
    const float* __restrict__ Wi,   // [2,1536,512]
    const float* __restrict__ Wh,   // [2,1536,512]
    const float* __restrict__ lnw,  // [2,3,512]
    const float* __restrict__ lnb,  // [2,3,512]
    float* __restrict__ out,        // [16,1500,512]
    unsigned* __restrict__ cnt,
    float* __restrict__ hf,            // [2,16,512] f32 hidden state
    unsigned short* __restrict__ hb,   // [2,16,512] bf16 hidden state
    float* __restrict__ gx,            // [16,1536] i2h gate staging
    float* __restrict__ gh)            // [16,1536] h2h gate staging
{
  extern __shared__ char smem[];
  unsigned*        Wpack = (unsigned*)(smem + LDS_WPACK);
  unsigned short*  Abuf  = (unsigned short*)(smem + LDS_ABUF);
  float*           part  = (float*)(smem + LDS_PART);
  float2*          red   = (float2*)(smem + LDS_RED);

  const int  tid  = threadIdx.x;
  const int  lane = tid & 31;
  const int  wave = tid >> 5;
  const bool isWi = blockIdx.x < 32;            // blocks 0..31: i2h, 32..63: h2h
  const int  sliceBase = (blockIdx.x & 31) * SLICE_COLS;
  const float* Wsrc = isWi ? Wi : Wh;

  // ---- one-time: pack this block's weight slice (both layers) into LDS in
  //      WMMA bf16 B-matrix layout: [l*3+tile][kchunk][lane][8 dwords]
  for (int s = tid; s < 2 * NTILES * 16 * 32 * 8; s += NTHREADS) {
    const int j  = s & 7;
    const int L  = (s >> 3) & 31;
    const int c  = (s >> 8) & 15;
    const int lt = s >> 12;                      // l*3 + tile
    const int n  = sliceBase + (lt % 3) * 16 + (L & 15);
    const int l  = lt / 3;
    const int base = (L & 16) ? 8 : 0;
    const int k  = c * 32 + ((j < 4) ? (base + 2 * j) : (16 + base + 2 * (j - 4)));
    const float* wr = Wsrc + ((size_t)l * NGATE + n) * HDIM;
    const unsigned lo = f32_to_bf16(wr[k]);
    const unsigned hi = f32_to_bf16(wr[k + 1]);
    Wpack[s] = lo | (hi << 16);
  }

  unsigned bi = 0;
  for (int t = 0; t < T_STEPS; ++t) {
    for (int l = 0; l < 2; ++l) {
      __syncthreads();
      // ---- stage this block's A matrix (16x512, bf16) into LDS
      if (isWi && l == 0) {
        // A = x[:, t, :]  (f32 -> bf16), float4 loads / uint2 LDS stores
        for (int s = tid; s < 2048; s += NTHREADS) {
          const int b = s >> 7;
          const int q = s & 127;
          const float4 xv =
              *(const float4*)(x + ((size_t)(b * T_STEPS + t)) * HDIM + q * 4);
          uint2 pk;
          pk.x = (unsigned)f32_to_bf16(xv.x) | ((unsigned)f32_to_bf16(xv.y) << 16);
          pk.y = (unsigned)f32_to_bf16(xv.z) | ((unsigned)f32_to_bf16(xv.w) << 16);
          ((uint2*)Abuf)[s] = pk;
        }
      } else {
        // Wi blocks, l=1: A = h0_new;  Wh blocks: A = h[l]  -- TDM DMA copy
        if (wave == 0) {
          const unsigned short* src = hb + (size_t)(isWi ? (l - 1) : l) * (BATCH * HDIM);
          tdm_copy_to_lds(src, LDS_ABUF, BATCH * HDIM);
          __builtin_amdgcn_s_wait_tensorcnt(0);
        }
      }
      __syncthreads();

      // ---- GEMM: [16x512]bf16 x LDS-resident [512x48]bf16, split-K over 6 waves
      if (wave < 6) {
        const int tl = wave % 3;      // N tile
        const int kh = wave / 3;      // K half
        v8f acc = {};
        #pragma unroll
        for (int cc = 0; cc < 8; ++cc) {
          const int c = kh * 8 + cc;
          Frag16 fa;
          {
            const int row  = lane & 15;
            const int base = (lane & 16) ? 8 : 0;
            const unsigned short* p = Abuf + row * HDIM + c * 32 + base;
            fa.q[0] = *(const uint4*)(p);        // ds_load_b128
            fa.q[1] = *(const uint4*)(p + 16);   // ds_load_b128
          }
          Frag16 fb;
          {
            const unsigned* p = Wpack + (((l * 3 + tl) * 16 + c) << 8) + lane * 8;
            fb.q[0] = *(const uint4*)(p);        // ds_load_b128
            fb.q[1] = *(const uint4*)(p + 4);    // ds_load_b128
          }
          acc = __builtin_amdgcn_wmma_f32_16x16x32_bf16(
                    false, fa.v, false, fb.v, (short)0, acc, false, false);
        }
        // spill split-K partials as two b128 stores
        float* pp = part + (wave * 32 + lane) * 8;
        *(float4*)(pp)     = make_float4(acc[0], acc[1], acc[2], acc[3]);
        *(float4*)(pp + 4) = make_float4(acc[4], acc[5], acc[6], acc[7]);
      }
      __syncthreads();
      if (wave < 3) {
        const float* p0 = part + (wave * 32 + lane) * 8;
        const float* p1 = part + ((wave + 3) * 32 + lane) * 8;
        const float4 a0 = *(const float4*)(p0);       // ds_load_b128
        const float4 a1 = *(const float4*)(p0 + 4);
        const float4 b0 = *(const float4*)(p1);
        const float4 b1 = *(const float4*)(p1 + 4);
        const float  rv[8] = {a0.x + b0.x, a0.y + b0.y, a0.z + b0.z, a0.w + b0.w,
                              a1.x + b1.x, a1.y + b1.y, a1.z + b1.z, a1.w + b1.w};
        const int colg = sliceBase + wave * 16 + (lane & 15);
        const int mb   = (lane >> 4) << 3;      // C/D layout: lanes 16..31 -> M+8
        float* g = isWi ? gx : gh;
        #pragma unroll
        for (int v = 0; v < 8; ++v)
          g[(size_t)(mb + v) * NGATE + colg] = rv[v];
      }

      grid_barrier(cnt, bi);

      // ---- LayerNorm + gates + state update: block b handles batch row b
      if (blockIdx.x < BATCH) {
        const int b = blockIdx.x;
        const float* gxr = gx + (size_t)b * NGATE;
        const float* ghr = gh + (size_t)b * NGATE;
        const int i0 = tid, i1 = tid + 256;
        float r0, r1, z0, z1, n0, n1;
        {   // r = sigmoid(LN(x_r + h_r))
          const float u0 = gxr[i0] + ghr[i0];
          const float u1 = gxr[i1] + ghr[i1];
          const float2 sq = block_reduce2(u0 + u1, u0 * u0 + u1 * u1, red);
          const float mean = sq.x * (1.f / 512.f);
          const float rstd = rsqrtf(sq.y * (1.f / 512.f) - mean * mean + 1e-5f);
          const float* w  = lnw + (l * 3 + 0) * HDIM;
          const float* bb = lnb + (l * 3 + 0) * HDIM;
          r0 = sigmf((u0 - mean) * rstd * w[i0] + bb[i0]);
          r1 = sigmf((u1 - mean) * rstd * w[i1] + bb[i1]);
        }
        {   // z = sigmoid(LN(x_z + h_z))
          const float u0 = gxr[512 + i0] + ghr[512 + i0];
          const float u1 = gxr[512 + i1] + ghr[512 + i1];
          const float2 sq = block_reduce2(u0 + u1, u0 * u0 + u1 * u1, red);
          const float mean = sq.x * (1.f / 512.f);
          const float rstd = rsqrtf(sq.y * (1.f / 512.f) - mean * mean + 1e-5f);
          const float* w  = lnw + (l * 3 + 1) * HDIM;
          const float* bb = lnb + (l * 3 + 1) * HDIM;
          z0 = sigmf((u0 - mean) * rstd * w[i0] + bb[i0]);
          z1 = sigmf((u1 - mean) * rstd * w[i1] + bb[i1]);
        }
        {   // n = tanh(LN(x_n + r*h_n))
          const float u0 = gxr[1024 + i0] + r0 * ghr[1024 + i0];
          const float u1 = gxr[1024 + i1] + r1 * ghr[1024 + i1];
          const float2 sq = block_reduce2(u0 + u1, u0 * u0 + u1 * u1, red);
          const float mean = sq.x * (1.f / 512.f);
          const float rstd = rsqrtf(sq.y * (1.f / 512.f) - mean * mean + 1e-5f);
          const float* w  = lnw + (l * 3 + 2) * HDIM;
          const float* bb = lnb + (l * 3 + 2) * HDIM;
          n0 = tanhf((u0 - mean) * rstd * w[i0] + bb[i0]);
          n1 = tanhf((u1 - mean) * rstd * w[i1] + bb[i1]);
        }
        float* hrow = hf + ((size_t)l * BATCH + b) * HDIM;
        unsigned short* hbrow = hb + ((size_t)l * BATCH + b) * HDIM;
        float hn0 = (1.f - z0) * n0 + z0 * hrow[i0];
        float hn1 = (1.f - z1) * n1 + z1 * hrow[i1];
        if (l == 1) {                 // residual from layer-0 output (this t)
          const float* inr = hf + (size_t)b * HDIM;
          hn0 += inr[i0];
          hn1 += inr[i1];
          float* orow = out + ((size_t)b * T_STEPS + t) * HDIM;
          orow[i0] = hn0;
          orow[i1] = hn1;
        }
        hrow[i0] = hn0;          hrow[i1] = hn1;
        hbrow[i0] = f32_to_bf16(hn0);
        hbrow[i1] = f32_to_bf16(hn1);
      }

      grid_barrier(cnt, bi);
    }
  }
}

extern "C" void kernel_launch(void* const* d_in, const int* in_sizes, int n_in,
                              void* d_out, int out_size, void* d_ws, size_t ws_size,
                              hipStream_t stream) {
  (void)in_sizes; (void)n_in; (void)out_size; (void)ws_size;
  const float* x   = (const float*)d_in[0];
  const float* Wi  = (const float*)d_in[1];
  const float* Wh  = (const float*)d_in[2];
  const float* lnw = (const float*)d_in[3];
  const float* lnb = (const float*)d_in[4];
  float* out = (float*)d_out;

  // workspace layout (~295 KB used)
  char* ws = (char*)d_ws;
  unsigned*       cnt = (unsigned*)ws;                       // barrier counter
  float*          hf  = (float*)(ws + 256);                  // 65536 B
  unsigned short* hb  = (unsigned short*)(ws + 256 + 65536); // 32768 B
  float*          gx  = (float*)(ws + 256 + 65536 + 32768);  // 98304 B
  float*          gh  = gx + (size_t)BATCH * NGATE;          // 98304 B

  gru_init_39745627357265<<<16, 256, 0, stream>>>(cnt, hf, hb);
  gru_persistent_39745627357265<<<NBLK, NTHREADS, LDS_BYTES, stream>>>(
      x, Wi, Wh, lnw, lnb, out, cnt, hf, hb, gx, gh);
}
